// SwinTransformer3D_63075889709440
// MI455X (gfx1250) — matrix operand order repaired and checked by reading, not compiled, not run on hardware
//
#include <hip/hip_runtime.h>
#include <cstddef>

typedef __attribute__((ext_vector_type(16))) _Float16 v16h;
typedef __attribute__((ext_vector_type(8)))  _Float16 v8h;
typedef __attribute__((ext_vector_type(8)))  float    v8f;
typedef __attribute__((ext_vector_type(4)))  float    v4f;

#define NWIN 98
#define NPAD 112
#define HD   24
#define BM 64
#define BN 64
#define BK 32

__device__ __forceinline__ v8h cvt8(v4f a, v4f b) {
  v8h h;
  h[0] = (_Float16)a[0]; h[1] = (_Float16)a[1];
  h[2] = (_Float16)a[2]; h[3] = (_Float16)a[3];
  h[4] = (_Float16)b[0]; h[5] = (_Float16)b[1];
  h[6] = (_Float16)b[2]; h[7] = (_Float16)b[3];
  return h;
}
__device__ __forceinline__ v16h cat16(v8h lo, v8h hi) {
  return __builtin_shufflevector(lo, hi, 0, 1, 2, 3, 4, 5, 6, 7,
                                 8, 9, 10, 11, 12, 13, 14, 15);
}
// A fragment from a row-major f16 row (stride must keep 16B alignment):
// lane needs halves [8*hi..8*hi+7] and [16+8*hi..23+8*hi]  -> 2x ds_load_b128
__device__ __forceinline__ v16h a_frag(const _Float16* rowbase, int hi) {
  const v8h* p = (const v8h*)rowbase;
  return cat16(p[hi], p[2 + hi]);
}
// B fragment from transposed storage: lane needs halves [8*hi..8*hi+7] and [16+8*hi..23+8*hi]
__device__ __forceinline__ v16h b_frag(const _Float16* rowbase, int hi) {
  const v8h* p = (const v8h*)rowbase;
  return cat16(p[hi], p[2 + hi]);
}

// ---------------------------------------------------------------------------
// Patch embedding: x (B,1,D,H,W) -> tokens (B*Dp*Hp*Wp, 96), patch (2,4,4)
// ---------------------------------------------------------------------------
__global__ void __launch_bounds__(256) patch_embed_kernel(
    const float* __restrict__ x, const float* __restrict__ w,
    const float* __restrict__ bias, float* __restrict__ out,
    int B, int Dp, int Hp, int Wp) {
  size_t idx = (size_t)blockIdx.x * 256 + threadIdx.x;
  size_t total = (size_t)B * Dp * Hp * Wp * 96;
  if (idx >= total) return;
  int co = (int)(idx % 96);
  size_t tok = idx / 96;
  int wp = (int)(tok % Wp); size_t t2 = tok / Wp;
  int hp = (int)(t2 % Hp);  size_t t3 = t2 / Hp;
  int dp = (int)(t3 % Dp);  int b = (int)(t3 / Dp);
  int Df = Dp * 2, Hf = Hp * 4, Wf = Wp * 4;
  const float* xb = x + (size_t)b * Df * Hf * Wf;
  float acc = bias[co];
#pragma unroll
  for (int z = 0; z < 2; z++)
#pragma unroll
    for (int yy = 0; yy < 4; yy++)
#pragma unroll
      for (int xx = 0; xx < 4; xx++) {
        float v = xb[(size_t)(dp * 2 + z) * Hf * Wf + (size_t)(hp * 4 + yy) * Wf + (wp * 4 + xx)];
        acc += w[co * 32 + z * 16 + yy * 4 + xx] * v;
      }
  out[idx] = acc;
}

// ---------------------------------------------------------------------------
// Row LayerNorm: one block per row
// ---------------------------------------------------------------------------
__global__ void __launch_bounds__(256) ln_kernel(
    const float* __restrict__ in, const float* __restrict__ g,
    const float* __restrict__ b, float* __restrict__ out, int C) {
  int row = blockIdx.x;
  const float* p = in + (size_t)row * C;
  float s = 0.f, s2 = 0.f;
  for (int c = threadIdx.x; c < C; c += 256) { float v = p[c]; s += v; s2 += v * v; }
  __shared__ float red0[256], red1[256];
  red0[threadIdx.x] = s; red1[threadIdx.x] = s2;
  __syncthreads();
  for (int off = 128; off > 0; off >>= 1) {
    if ((int)threadIdx.x < off) {
      red0[threadIdx.x] += red0[threadIdx.x + off];
      red1[threadIdx.x] += red1[threadIdx.x + off];
    }
    __syncthreads();
  }
  float mean = red0[0] / C;
  float var  = red1[0] / C - mean * mean;
  float inv  = rsqrtf(var + 1e-5f);
  for (int c = threadIdx.x; c < C; c += 256)
    out[(size_t)row * C + c] = (p[c] - mean) * inv * g[c] + b[c];
}

// ---------------------------------------------------------------------------
// WMMA GEMM: out[m,n] = act(A[m,:] . W[n,:] + bias[n]) (+ resid[m,n])
// A: [M,K] f32, W: [N,K] f32 row-major (implicit W^T). K % 32 == 0 for all
// call sites. Block = 64x64 tile, 8 waves, each wave -> 1x2 16x16 tiles
// sharing one A fragment (2 wmma per staging round).
// ---------------------------------------------------------------------------
__global__ void __launch_bounds__(256) gemm_kernel(
    const float* __restrict__ A, const float* __restrict__ Wt,
    const float* __restrict__ bias, const float* __restrict__ resid,
    float* __restrict__ out, int M, int N, int K, int act) {
  __shared__ _Float16 As[BM][BK + 8];    // row stride 80B (16B multiple)
  __shared__ _Float16 BsT[BN][BK + 8];   // transposed: [n][k]
  int tid = threadIdx.x;
  int wave = tid >> 5, lane = tid & 31;
  int hi = lane >> 4, nl = lane & 15;
  int wm = wave & 3, wn = wave >> 2;     // 4 (M) x 2 (N) waves
  int tileM0 = blockIdx.x * BM;
  int tileN0 = blockIdx.y * BN;
  int r  = tid >> 2;                      // staging row 0..63
  int c0 = (tid & 3) * 8;                 // staging K chunk
  v8f acc0 = {}, acc1 = {};
  for (int k0 = 0; k0 < K; k0 += BK) {
    {
      int gm = tileM0 + r;
      v8h h = {};
      if (gm < M) {
        const float* src = &A[(size_t)gm * K + k0 + c0];
        h = cvt8(*(const v4f*)src, *(const v4f*)(src + 4));
      }
      *(v8h*)&As[r][c0] = h;
    }
    {
      int gn = tileN0 + r;
      v8h h = {};
      if (gn < N) {
        const float* src = &Wt[(size_t)gn * K + k0 + c0];
        h = cvt8(*(const v4f*)src, *(const v4f*)(src + 4));
      }
      *(v8h*)&BsT[r][c0] = h;
    }
    if (k0 + BK < K) {
      int pr = tileM0 + r;
      if (pr < M) __builtin_prefetch(&A[(size_t)pr * K + k0 + BK], 0, 1);
    }
    __syncthreads();
    v16h a  = a_frag(&As[wm * 16 + nl][0], hi);
    v16h b0 = b_frag(&BsT[wn * 32 + nl][0], hi);
    v16h b1 = b_frag(&BsT[wn * 32 + 16 + nl][0], hi);
    acc0 = __builtin_amdgcn_wmma_f32_16x16x32_f16(false, a, false, b0,
                                                  (short)0, acc0, false, false);
    acc1 = __builtin_amdgcn_wmma_f32_16x16x32_f16(false, a, false, b1,
                                                  (short)0, acc1, false, false);
    __syncthreads();
  }
#pragma unroll
  for (int rr = 0; rr < 8; rr++) {
    int m = tileM0 + wm * 16 + rr + hi * 8;
    if (m >= M) continue;
#pragma unroll
    for (int t = 0; t < 2; t++) {
      int n = tileN0 + wn * 32 + t * 16 + nl;
      if (n >= N) continue;
      float v = (t == 0) ? acc0[rr] : acc1[rr];
      if (bias)  v += bias[n];
      if (act == 1) v = 0.5f * v * (1.0f + erff(v * 0.70710678118654752f));
      if (resid) v += resid[(size_t)m * N + n];
      out[(size_t)m * N + n] = v;
    }
  }
}

// ---------------------------------------------------------------------------
// Fused window attention, one block per (window, head). head_dim == 24.
// QK^T (WMMA) -> +rel-pos bias -> softmax in LDS fp32 -> P.V (WMMA).
// Shift handled by modular gather/scatter (cyclic roll, no mask — matches ref).
// ---------------------------------------------------------------------------
__device__ __forceinline__ int win_token_row(int t, int b, int wd, int wh, int ww,
                                             int D, int H, int W,
                                             int sd, int sh, int sw) {
  int z = t / 49, y = (t / 7) % 7, x = t % 7;
  int gd = (wd * 2 + z + sd) % D;
  int gh = (wh * 7 + y + sh) % H;
  int gw = (ww * 7 + x + sw) % W;
  return b * (D * H * W) + (gd * H + gh) * W + gw;
}

__global__ void __launch_bounds__(256) attn_kernel(
    const float* __restrict__ qkv,   // [T, 3C]
    const float* __restrict__ rpb,   // [507, nh]
    float* __restrict__ attn_out,    // [T, C]
    int D, int H, int W, int C, int nh, int sd, int sh, int sw) {
  int win = blockIdx.x, head = blockIdx.y;
  int nW = W / 7, nH = H / 7, nD = D / 2;
  int b  = win / (nD * nH * nW);
  int rr = win % (nD * nH * nW);
  int wd = rr / (nH * nW);
  int wh = (rr / nW) % nH;
  int ww = rr % nW;
  int C3 = 3 * C;

  __shared__ _Float16 qs [NPAD][32];   // A of S: [token][k], rows 64B
  __shared__ _Float16 ksT[NPAD][32];   // B of S transposed: [token(n)][k]
  __shared__ _Float16 vsT[32][NPAD];   // B of O transposed: [d(n)][token(k)]
  __shared__ float    S  [NWIN][NPAD]; // scores -> probabilities

  int tid = threadIdx.x;
  float scale = rsqrtf((float)HD);

  // staging: thread handles one (token, 8-wide d chunk); 112*4 = 448 units
  for (int i = tid; i < NPAD * 4; i += 256) {
    int t = i >> 2, d0 = (i & 3) * 8;
    v8h hq = {}, hk = {}, hv = {};
    if (t < NWIN && d0 < HD) {
      int row = win_token_row(t, b, wd, wh, ww, D, H, W, sd, sh, sw);
      const float* qp = qkv + (size_t)row * C3 + head * HD + d0;  // 32B aligned
      v4f q0 = *(const v4f*)qp, q1 = *(const v4f*)(qp + 4);
      q0 *= scale; q1 *= scale;
      hq = cvt8(q0, q1);
      hk = cvt8(*(const v4f*)(qp + C), *(const v4f*)(qp + C + 4));
      hv = cvt8(*(const v4f*)(qp + 2 * C), *(const v4f*)(qp + 2 * C + 4));
    }
    *(v8h*)&qs[t][d0]  = hq;
    *(v8h*)&ksT[t][d0] = hk;
#pragma unroll
    for (int j = 0; j < 8; j++) vsT[d0 + j][t] = hv[j];
  }
  __syncthreads();

  int wave = tid >> 5, lane = tid & 31, hi = lane >> 4, nl = lane & 15;

  // S = q k^T : 7x7 tiles, K = 24 zero-padded to 32
  for (int tile = wave; tile < 49; tile += 8) {
    int tm = tile / 7, tn = tile % 7;
    v16h a = a_frag(&qs[tm * 16 + nl][0], hi);
    v16h bb = b_frag(&ksT[tn * 16 + nl][0], hi);
    v8f c = {};
    c = __builtin_amdgcn_wmma_f32_16x16x32_f16(false, a, false, bb,
                                               (short)0, c, false, false);
#pragma unroll
    for (int r = 0; r < 8; r++) {
      int m = tm * 16 + r + hi * 8;
      int n = tn * 16 + nl;
      if (m < NWIN) S[m][n] = c[r];
    }
  }
  __syncthreads();

  // bias + softmax, one thread per row
  for (int i = tid; i < NWIN; i += 256) {
    int zi = i / 49, yi = (i / 7) % 7, xi = i % 7;
    float mx = -1e30f;
    for (int j = 0; j < NWIN; j++) {
      int zj = j / 49, yj = (j / 7) % 7, xj = j % 7;
      int idx = (zi - zj + 1) * 169 + (yi - yj + 6) * 13 + (xi - xj + 6);
      float sv = S[i][j] + rpb[idx * nh + head];
      S[i][j] = sv;
      mx = fmaxf(mx, sv);
    }
    float sum = 0.f;
    for (int j = 0; j < NWIN; j++) { float e = __expf(S[i][j] - mx); S[i][j] = e; sum += e; }
    float inv = 1.f / sum;
    for (int j = 0; j < NWIN; j++) S[i][j] *= inv;
    for (int j = NWIN; j < NPAD; j++) S[i][j] = 0.f;   // zero K-pads for P.V
  }
  __syncthreads();

  // O = P v : 7 M-tiles x 2 N-tiles, K = 112 (zeros past 98) in 4 steps
  for (int tile = wave; tile < 14; tile += 8) {
    int tm = tile / 2, tn = tile % 2;
    v8f c = {};
    int ar = tm * 16 + nl; if (ar >= NWIN) ar = NWIN - 1;  // clamp M pad rows
#pragma unroll
    for (int k0 = 0; k0 < NPAD; k0 += 32) {
      // A fragment: two 8-float runs of row ar, converted f32->f16
      int base1 = k0 + hi * 16;
      int base2 = k0 + hi * 16 + 8;
      v4f f0 = {}, f1 = {}, f2 = {}, f3 = {};
      if (base1 < NPAD) {
        const float* p = &S[ar][base1];
        f0 = *(const v4f*)p; f1 = *(const v4f*)(p + 4);
      }
      if (base2 < NPAD) {
        const float* p = &S[ar][base2];
        f2 = *(const v4f*)p; f3 = *(const v4f*)(p + 4);
      }
      v16h a = cat16(cvt8(f0, f1), cvt8(f2, f3));
      // B fragment: contiguous token run of vsT row (d = tn*16+nl)
      int kb = k0 + hi * 16;
      v8h b0 = {}, b1 = {};
      if (kb < NPAD) {
        const v8h* p = (const v8h*)&vsT[tn * 16 + nl][kb];
        b0 = p[0]; b1 = p[1];
      }
      v16h bb = cat16(b0, b1);
      c = __builtin_amdgcn_wmma_f32_16x16x32_f16(false, a, false, bb,
                                                 (short)0, c, false, false);
    }
#pragma unroll
    for (int r = 0; r < 8; r++) {
      int m = tm * 16 + r + hi * 8;
      int n = tn * 16 + nl;
      if (m < NWIN && n < HD) {
        int row = win_token_row(m, b, wd, wh, ww, D, H, W, sd, sh, sw);
        attn_out[(size_t)row * C + head * HD + n] = c[r];
      }
    }
  }
}

// ---------------------------------------------------------------------------
// Patch-merge gather: X[B*D*H*W, C] -> out[B*(D/2)(H/2)(W/2), 8C]
// ---------------------------------------------------------------------------
__global__ void __launch_bounds__(256) merge_gather_kernel(
    const float* __restrict__ X, float* __restrict__ out,
    int B, int D, int H, int W, int C) {
  int D2 = D / 2, H2 = H / 2, W2 = W / 2;
  size_t total = (size_t)B * D2 * H2 * W2 * 8 * C;
  size_t idx = (size_t)blockIdx.x * 256 + threadIdx.x;
  if (idx >= total) return;
  const int oz[8] = {0, 1, 0, 0, 1, 1, 0, 1};
  const int oy[8] = {0, 0, 1, 0, 1, 0, 1, 1};
  const int ox[8] = {0, 0, 0, 1, 0, 1, 1, 1};
  int C8 = 8 * C;
  int c8 = (int)(idx % C8); size_t tok = idx / C8;
  int g = c8 / C, c = c8 % C;
  int xw = (int)(tok % W2); size_t t2 = tok / W2;
  int yh = (int)(t2 % H2);  size_t t3 = t2 / H2;
  int zd = (int)(t3 % D2);  int b = (int)(t3 / D2);
  size_t srcrow = ((size_t)b * D + (zd * 2 + oz[g])) * H * W
                + (size_t)(yh * 2 + oy[g]) * W + (xw * 2 + ox[g]);
  out[idx] = X[srcrow * C + c];
}

// ---------------------------------------------------------------------------
// Host orchestration
// ---------------------------------------------------------------------------
extern "C" void kernel_launch(void* const* d_in, const int* in_sizes, int n_in,
                              void* d_out, int out_size, void* d_ws, size_t ws_size,
                              hipStream_t stream) {
  (void)in_sizes; (void)n_in; (void)out_size; (void)ws_size;
  const float* xin   = (const float*)d_in[0];
  const float* emb_w = (const float*)d_in[1];
  const float* emb_b = (const float*)d_in[2];

  float* ws = (float*)d_ws;
  const size_t XCAP = (size_t)2 * 16 * 56 * 56 * 96;  // stage-0 token block
  float* X   = ws;
  float* TMP = ws + XCAP;
  float* ATT = ws + 2 * XCAP;
  float* BIG = ws + 3 * XCAP;                          // 4*XCAP floats
  float* OUT = (float*)d_out;

  {
    size_t total = (size_t)2 * 16 * 56 * 56 * 96;
    patch_embed_kernel<<<dim3((unsigned)((total + 255) / 256)), 256, 0, stream>>>(
        xin, emb_w, emb_b, X, 2, 16, 56, 56);
  }

  auto gemm = [&](const float* A, const float* Wm, const float* bias,
                  const float* resid, float* out, int M, int N, int K, int act) {
    dim3 grid((M + BM - 1) / BM, (N + BN - 1) / BN);
    gemm_kernel<<<grid, 256, 0, stream>>>(A, Wm, bias, resid, out, M, N, K, act);
  };

  const int dims[3]  = {96, 192, 384};
  const int heads[3] = {4, 8, 16};
  int D = 16, H = 56, W = 56;
  int base = 3;

  for (int s = 0; s < 3; s++) {
    int C = dims[s], nh = heads[s];
    int T = 2 * D * H * W;
    for (int blk = 0; blk < 2; blk++) {
      const float* const* p = (const float* const*)(d_in + base + blk * 13);
      const float *n1g = p[0], *n1b = p[1], *qkvw = p[2], *qkvb = p[3], *rpb = p[4];
      const float *projw = p[5], *projb = p[6], *n2g = p[7], *n2b = p[8];
      const float *fc1w = p[9], *fc1b = p[10], *fc2w = p[11], *fc2b = p[12];
      int sd = (blk & 1) ? 1 : 0, sh = (blk & 1) ? 3 : 0, sw = (blk & 1) ? 3 : 0;

      ln_kernel<<<T, 256, 0, stream>>>(X, n1g, n1b, TMP, C);
      gemm(TMP, qkvw, qkvb, nullptr, BIG, T, 3 * C, C, 0);
      {
        int wins = 2 * (D / 2) * (H / 7) * (W / 7);
        attn_kernel<<<dim3(wins, nh), 256, 0, stream>>>(BIG, rpb, ATT,
                                                        D, H, W, C, nh, sd, sh, sw);
      }
      gemm(ATT, projw, projb, X, X, T, C, C, 0);
      ln_kernel<<<T, 256, 0, stream>>>(X, n2g, n2b, TMP, C);
      gemm(TMP, fc1w, fc1b, nullptr, BIG, T, 4 * C, C, 1);
      bool last = (s == 2 && blk == 1);
      gemm(BIG, fc2w, fc2b, X, last ? OUT : X, T, C, 4 * C, 0);
    }
    base += 26;
    if (s < 2) {
      const float* ng   = (const float*)d_in[base + 0];
      const float* nb   = (const float*)d_in[base + 1];
      const float* redw = (const float*)d_in[base + 2];
      base += 3;
      int Tn = 2 * (D / 2) * (H / 2) * (W / 2);
      size_t total = (size_t)Tn * 8 * C;
      merge_gather_kernel<<<dim3((unsigned)((total + 255) / 256)), 256, 0, stream>>>(
          X, ATT, 2, D, H, W, C);
      ln_kernel<<<Tn, 256, 0, stream>>>(ATT, ng, nb, TMP, 8 * C);
      gemm(TMP, redw, nullptr, nullptr, X, Tn, 2 * C, 8 * C, 0);
      D /= 2; H /= 2; W /= 2;
    }
  }
}
